// Mamba2Discriminator_33432025432259
// MI455X (gfx1250) — matrix operand order repaired and checked by reading, not compiled, not run on hardware
//
#include <hip/hip_runtime.h>
#include <hip/hip_bf16.h>

#define BATCH   128
#define MAXLEN  128
#define D_MODEL 512
#define D_INNER 1024
#define D_STATE 128
#define D_CONV  4
#define HEADDIM 64
#define NHEADS  16
#define NLAYERS 2
#define CONV_DIM 1280            // D_INNER + 2*D_STATE
#define D_IN_PROJ 2320           // 2*D_INNER + 2*D_STATE + NHEADS
#define D_IN_PROJ_PAD 2368       // padded to multiple of 64 for guard-free WMMA tiles
#define FC_HID  128
#define EPS_F   1e-5f
#define ROWS    (BATCH * MAXLEN) // 16384

typedef __attribute__((ext_vector_type(16))) __bf16 v16bf;
typedef __attribute__((ext_vector_type(8)))  __bf16 v8bf;
typedef __attribute__((ext_vector_type(8)))  float  v8f;

#define CAT16(lo, hi) __builtin_shufflevector(lo, hi, 0,1,2,3,4,5,6,7,8,9,10,11,12,13,14,15)

// ---------------------------------------------------------------------------
// f32 -> bf16 weight conversion with zero row padding:
// dst[(layer*Npad + n)*K + k] = n < N ? bf16(src[(layer*N + n)*K + k]) : 0
// ---------------------------------------------------------------------------
__global__ __launch_bounds__(256) void cvt_pad_bf16_kernel(const float* __restrict__ src,
                                                           __hip_bfloat16* __restrict__ dst,
                                                           int N, int Npad, int K) {
  size_t i = (size_t)blockIdx.x * 256 + threadIdx.x;
  size_t total = (size_t)NLAYERS * Npad * K;
  if (i >= total) return;
  int    k     = (int)(i % K);
  size_t nl    = i / K;
  int    n     = (int)(nl % Npad);
  int    layer = (int)(nl / Npad);
  float v = (n < N) ? src[((size_t)layer * N + n) * K + k] : 0.f;
  dst[i] = __float2bfloat16(v);
}

// ---------------------------------------------------------------------------
// Embedding + positional: h[b,l,:] = emb[tok[b,l],:] + pos[l,:]  (f32 + bf16)
// ---------------------------------------------------------------------------
__global__ __launch_bounds__(256) void embed_kernel(const int* __restrict__ tok,
                                                    const float* __restrict__ emb,
                                                    const float* __restrict__ pos,
                                                    float* __restrict__ h,
                                                    __hip_bfloat16* __restrict__ hbf) {
  int row = blockIdx.x;              // b*L + l
  int t   = threadIdx.x;
  int l   = row & (MAXLEN - 1);
  int tk  = tok[row];
#pragma unroll
  for (int i = 0; i < 2; ++i) {
    int c = t + i * 256;
    float v = emb[(size_t)tk * D_MODEL + c] + pos[(size_t)l * D_MODEL + c];
    h[(size_t)row * D_MODEL + c]   = v;
    hbf[(size_t)row * D_MODEL + c] = __float2bfloat16(v);
  }
}

// ---------------------------------------------------------------------------
// WMMA bf16 GEMM:  C[M, Npad] (f32) = A[M,K] (bf16) @ W[Npad,K]^T (bf16)
// Npad is a multiple of 64 (weights zero-padded) -> no guards, EXEC all-1s.
// One wave computes a 32x64 tile: 2 A-frags x 4 B-frags -> 8 WMMAs / K-step.
// ---------------------------------------------------------------------------
__global__ __launch_bounds__(256) void gemm_bf16_wmma(const __bf16* __restrict__ A,
                                                      const __bf16* __restrict__ W,
                                                      float* __restrict__ C,
                                                      int M, int Npad, int K) {
  const int lane   = threadIdx.x & 31;
  const int wave   = threadIdx.x >> 5;
  const int tilesN = Npad >> 6;
  const int gw     = blockIdx.x * 8 + wave;
  const int tm     = gw / tilesN;
  const int tn     = gw - tm * tilesN;
  if (tm * 32 >= M) return;                 // wave-uniform

  const int kbA  = (lane >> 4) * 8;         // A: K sub-offset (ISA 16-bit A layout)
  const int kbB  = (lane >> 4) * 8;         // B: K sub-offset (ISA 16-bit B layout)
  const int nloc = lane & 15;
  const int n0   = tn * 64;

  v8f acc0[4] = {};
  v8f acc1[4] = {};
  const __bf16* Arow0 = A + (size_t)(tm * 32 + (lane & 15)) * K;
  const __bf16* Arow1 = Arow0 + (size_t)16 * K;

  for (int k0 = 0; k0 < K; k0 += 32) {
    v8bf a0lo = *(const v8bf*)(Arow0 + k0 + kbA);
    v8bf a0hi = *(const v8bf*)(Arow0 + k0 + 16 + kbA);
    v8bf a1lo = *(const v8bf*)(Arow1 + k0 + kbA);
    v8bf a1hi = *(const v8bf*)(Arow1 + k0 + 16 + kbA);
    v16bf a0 = CAT16(a0lo, a0hi);
    v16bf a1 = CAT16(a1lo, a1hi);

#pragma unroll
    for (int s = 0; s < 4; ++s) {
      const __bf16* Wrow = W + (size_t)(n0 + s * 16 + nloc) * K + k0 + kbB;
      v8bf blo = *(const v8bf*)(Wrow);
      v8bf bhi = *(const v8bf*)(Wrow + 16);
      v16bf b = CAT16(blo, bhi);
      acc0[s] = __builtin_amdgcn_wmma_f32_16x16x32_bf16(
          false, a0, false, b, (short)0, acc0[s], false, false);
      acc1[s] = __builtin_amdgcn_wmma_f32_16x16x32_bf16(
          false, a1, false, b, (short)0, acc1[s], false, false);
    }
  }

  const int rowHalf = (lane >> 4) * 8;      // C/D: M = r + 8*(lane>=16)
#pragma unroll
  for (int s = 0; s < 4; ++s) {
    int n = n0 + s * 16 + nloc;
    float* Cp0 = C + (size_t)(tm * 32 + rowHalf) * Npad + n;
    float* Cp1 = Cp0 + (size_t)16 * Npad;
#pragma unroll
    for (int r = 0; r < 8; ++r) {
      Cp0[(size_t)r * Npad] = acc0[s][r];
      Cp1[(size_t)r * Npad] = acc1[s][r];
    }
  }
}

// ---------------------------------------------------------------------------
// Depthwise causal conv over xBC (cols D_INNER..D_INNER+CONV_DIM of zxbcdt)
// + SiLU; also dt = softplus(raw_dt + dt_bias), dA = exp(-exp(A_log)*dt).
// One block per (b,l) row. zx rows have stride D_IN_PROJ_PAD.
// ---------------------------------------------------------------------------
__global__ __launch_bounds__(256) void conv_dt_kernel(const float* __restrict__ zx,
                                                      const float* __restrict__ cw,
                                                      const float* __restrict__ cb,
                                                      const float* __restrict__ dtb,
                                                      const float* __restrict__ alog,
                                                      float* __restrict__ xbcs,
                                                      float* __restrict__ dt,
                                                      float* __restrict__ dA) {
  int row = blockIdx.x;              // b*L + l
  int l   = row & (MAXLEN - 1);
  int t   = threadIdx.x;

  for (int c = t; c < CONV_DIM; c += 256) {
    float acc = cb[c];
#pragma unroll
    for (int k = 0; k < D_CONV; ++k) {
      int ls = l + k - (D_CONV - 1);
      float v = 0.f;
      if (ls >= 0)
        v = zx[(size_t)(row + k - (D_CONV - 1)) * D_IN_PROJ_PAD + D_INNER + c];
      acc += cw[c * D_CONV + k] * v;
    }
    float s = acc / (1.f + expf(-acc));     // SiLU
    xbcs[(size_t)row * CONV_DIM + c] = s;
  }
  if (t < NHEADS) {
    float v  = zx[(size_t)row * D_IN_PROJ_PAD + (D_IN_PROJ - NHEADS) + t] + dtb[t];
    float sp = (v > 20.f) ? v : log1pf(expf(v));
    dt[(size_t)row * NHEADS + t] = sp;
    dA[(size_t)row * NHEADS + t] = expf(-expf(alog[t]) * sp);
  }
}

// ---------------------------------------------------------------------------
// Recurrent SSM scan. Block = (batch, head). State 64x128 held in registers:
// thread t owns p = t>>2, n = (t&3)*32 + j, j in [0,32). y reduced via shfl.
// Writes y[b,l, h*64+p] = scan_y + D_skip[h]*x.
// ---------------------------------------------------------------------------
__global__ __launch_bounds__(256) void scan_kernel(const float* __restrict__ xbcs,
                                                   const float* __restrict__ dt,
                                                   const float* __restrict__ dA,
                                                   const float* __restrict__ dsk,
                                                   float* __restrict__ y) {
  int b = blockIdx.x >> 4;
  int h = blockIdx.x & (NHEADS - 1);
  int t = threadIdx.x;
  int p  = t >> 2;
  int nb = t & 3;

  __shared__ float xs[HEADDIM];
  __shared__ float Bs[D_STATE];
  __shared__ float Cs[D_STATE];
  __shared__ float sc[2];

  float st[32];
#pragma unroll
  for (int j = 0; j < 32; ++j) st[j] = 0.f;
  float dskh = dsk[h];

  for (int l = 0; l < MAXLEN; ++l) {
    size_t row = (size_t)b * MAXLEN + l;
    const float* xr = xbcs + row * CONV_DIM;
    __syncthreads();
    if (t < 64)       xs[t]       = xr[h * HEADDIM + t];
    if (t < 128)      Bs[t]       = xr[D_INNER + t];
    else              Cs[t - 128] = xr[D_INNER + D_STATE + (t - 128)];
    if (t == 0) { sc[0] = dt[row * NHEADS + h]; sc[1] = dA[row * NHEADS + h]; }
    __syncthreads();

    float dtv  = sc[0];
    float dav  = sc[1];
    float xp   = xs[p];
    float coef = dtv * xp;
    float yp   = 0.f;
#pragma unroll
    for (int j = 0; j < 32; ++j) {
      int n = nb * 32 + j;
      float s = st[j] * dav + coef * Bs[n];
      st[j] = s;
      yp += s * Cs[n];
    }
    yp += __shfl_xor(yp, 1, 32);
    yp += __shfl_xor(yp, 2, 32);
    if (nb == 0)
      y[row * D_INNER + h * HEADDIM + p] = yp + dskh * xp;
  }
}

// ---------------------------------------------------------------------------
// y = y * silu(z); RMS-normalize over D_INNER; * nrm_w; store bf16 for GEMM2.
// Block per (b,l). z read from padded zx rows.
// ---------------------------------------------------------------------------
__global__ __launch_bounds__(256) void gate_rms_kernel(const float* __restrict__ y,
                                                       const float* __restrict__ zx,
                                                       const float* __restrict__ nw,
                                                       __hip_bfloat16* __restrict__ ybf) {
  int row = blockIdx.x;
  int t   = threadIdx.x;
  const float* yr = y  + (size_t)row * D_INNER;
  const float* zr = zx + (size_t)row * D_IN_PROJ_PAD;   // z = first D_INNER cols

  float g[4];
  float ss = 0.f;
#pragma unroll
  for (int i = 0; i < 4; ++i) {
    int c = t + i * 256;
    float z  = zr[c];
    float gv = yr[c] * (z / (1.f + expf(-z)));
    g[i] = gv;
    ss += gv * gv;
  }
  __shared__ float red[256];
  red[t] = ss;
  __syncthreads();
  for (int s = 128; s > 0; s >>= 1) { if (t < s) red[t] += red[t + s]; __syncthreads(); }
  float r = rsqrtf(red[0] / (float)D_INNER + EPS_F);
#pragma unroll
  for (int i = 0; i < 4; ++i) {
    int c = t + i * 256;
    ybf[(size_t)row * D_INNER + c] = __float2bfloat16(g[i] * r * nw[c]);
  }
}

// ---------------------------------------------------------------------------
// h = h + LayerNorm(yproj); update f32 + bf16 copies. Block per (b,l).
// ---------------------------------------------------------------------------
__global__ __launch_bounds__(256) void resid_ln_kernel(const float* __restrict__ yproj,
                                                       const float* __restrict__ lnw,
                                                       const float* __restrict__ lnb,
                                                       float* __restrict__ h,
                                                       __hip_bfloat16* __restrict__ hbf) {
  int row = blockIdx.x;
  int t   = threadIdx.x;
  const float* yr = yproj + (size_t)row * D_MODEL;
  float v0 = yr[t], v1 = yr[t + 256];

  __shared__ float red[256];
  red[t] = v0 + v1;
  __syncthreads();
  for (int s = 128; s > 0; s >>= 1) { if (t < s) red[t] += red[t + s]; __syncthreads(); }
  float mean = red[0] / (float)D_MODEL;
  __syncthreads();
  float d0 = v0 - mean, d1 = v1 - mean;
  red[t] = d0 * d0 + d1 * d1;
  __syncthreads();
  for (int s = 128; s > 0; s >>= 1) { if (t < s) red[t] += red[t + s]; __syncthreads(); }
  float inv = rsqrtf(red[0] / (float)D_MODEL + EPS_F);

#pragma unroll
  for (int i = 0; i < 2; ++i) {
    int c = t + i * 256;
    float d  = (i == 0 ? d0 : d1);
    float ln = d * inv * lnw[c] + lnb[c];
    float hn = h[(size_t)row * D_MODEL + c] + ln;
    h[(size_t)row * D_MODEL + c]   = hn;
    hbf[(size_t)row * D_MODEL + c] = __float2bfloat16(hn);
  }
}

// ---------------------------------------------------------------------------
// Masked mean pool over L. Block per batch.
// ---------------------------------------------------------------------------
__global__ __launch_bounds__(256) void pool_kernel(const float* __restrict__ h,
                                                   float* __restrict__ pooled) {
  int b = blockIdx.x;
  int t = threadIdx.x;
  __shared__ float red[256];
  float acc0 = 0.f, acc1 = 0.f, cnt = 0.f;

  for (int l = 0; l < MAXLEN; ++l) {
    const float* hr = h + ((size_t)b * MAXLEN + l) * D_MODEL;
    float v0 = hr[t], v1 = hr[t + 256];
    red[t] = v0 + v1;
    __syncthreads();
    for (int s = 128; s > 0; s >>= 1) { if (t < s) red[t] += red[t + s]; __syncthreads(); }
    float m = (red[0] != 0.f) ? 1.f : 0.f;
    __syncthreads();
    acc0 += m * v0;
    acc1 += m * v1;
    cnt  += m;
  }
  float d = fmaxf(cnt, 1.f);
  pooled[(size_t)b * D_MODEL + t]       = acc0 / d;
  pooled[(size_t)b * D_MODEL + t + 256] = acc1 / d;
}

// ---------------------------------------------------------------------------
// Head: hid = relu(pooled @ fc1^T + b1); out = hid @ fc2^T + b2. Block per b.
// ---------------------------------------------------------------------------
__global__ __launch_bounds__(128) void head_kernel(const float* __restrict__ pooled,
                                                   const float* __restrict__ w1,
                                                   const float* __restrict__ b1,
                                                   const float* __restrict__ w2,
                                                   const float* __restrict__ b2,
                                                   float* __restrict__ out) {
  int b = blockIdx.x;
  int j = threadIdx.x;       // hidden unit
  const float* pr = pooled + (size_t)b * D_MODEL;
  const float* wr = w1 + (size_t)j * D_MODEL;
  float s = b1[j];
  for (int k = 0; k < D_MODEL; ++k) s += pr[k] * wr[k];
  float hid = fmaxf(s, 0.f);

  __shared__ float red[128];
  red[j] = hid * w2[j];
  __syncthreads();
  for (int st = 64; st > 0; st >>= 1) { if (j < st) red[j] += red[j + st]; __syncthreads(); }
  if (j == 0) out[b] = red[0] + b2[0];
}

// ---------------------------------------------------------------------------
// Host orchestration
// ---------------------------------------------------------------------------
static inline size_t align_up(size_t x) { return (x + 255) & ~(size_t)255; }

extern "C" void kernel_launch(void* const* d_in, const int* in_sizes, int n_in,
                              void* d_out, int out_size, void* d_ws, size_t ws_size,
                              hipStream_t stream) {
  const int*   tokens   = (const int*)  d_in[0];
  const float* emb      = (const float*)d_in[1];
  const float* pos      = (const float*)d_in[2];
  const float* in_w     = (const float*)d_in[3];
  const float* conv_w   = (const float*)d_in[4];
  const float* conv_b   = (const float*)d_in[5];
  const float* dt_bias  = (const float*)d_in[6];
  const float* A_log    = (const float*)d_in[7];
  const float* D_skip   = (const float*)d_in[8];
  const float* nrm_w    = (const float*)d_in[9];
  const float* out_w    = (const float*)d_in[10];
  const float* ln_w     = (const float*)d_in[11];
  const float* ln_b     = (const float*)d_in[12];
  const float* fc1_w    = (const float*)d_in[13];
  const float* fc1_b    = (const float*)d_in[14];
  const float* fc2_w    = (const float*)d_in[15];
  const float* fc2_b    = (const float*)d_in[16];
  float* out = (float*)d_out;

  // workspace carve-out
  char* ws = (char*)d_ws;
  size_t off = 0;
  auto take = [&](size_t bytes) -> char* { char* p = ws + off; off = align_up(off + bytes); return p; };

  const size_t n_w1_pad = (size_t)NLAYERS * D_IN_PROJ_PAD * D_MODEL;
  const size_t n_w2     = (size_t)NLAYERS * D_MODEL * D_INNER;

  __hip_bfloat16* W1bf = (__hip_bfloat16*)take(n_w1_pad * 2);
  __hip_bfloat16* W2bf = (__hip_bfloat16*)take(n_w2 * 2);
  float*          h    = (float*)take((size_t)ROWS * D_MODEL * 4);
  __hip_bfloat16* hbf  = (__hip_bfloat16*)take((size_t)ROWS * D_MODEL * 2);
  float*          zx   = (float*)take((size_t)ROWS * D_IN_PROJ_PAD * 4);
  float*          xbcs = (float*)take((size_t)ROWS * CONV_DIM * 4);
  float*          dtb_ = (float*)take((size_t)ROWS * NHEADS * 4);
  float*          dAb  = (float*)take((size_t)ROWS * NHEADS * 4);
  float*          ybuf = (float*)take((size_t)ROWS * D_INNER * 4);
  __hip_bfloat16* ybf  = (__hip_bfloat16*)take((size_t)ROWS * D_INNER * 2);
  float*          yprj = (float*)take((size_t)ROWS * D_MODEL * 4);
  float*          pooled = (float*)take((size_t)BATCH * D_MODEL * 4);
  (void)ws_size; (void)n_in; (void)in_sizes; (void)out_size;

  // weight conversion to bf16 (in_proj zero-padded to D_IN_PROJ_PAD rows)
  cvt_pad_bf16_kernel<<<(int)((n_w1_pad + 255) / 256), 256, 0, stream>>>(
      in_w, W1bf, D_IN_PROJ, D_IN_PROJ_PAD, D_MODEL);
  cvt_pad_bf16_kernel<<<(int)((n_w2 + 255) / 256), 256, 0, stream>>>(
      out_w, W2bf, D_MODEL, D_MODEL, D_INNER);

  // embedding
  embed_kernel<<<ROWS, 256, 0, stream>>>(tokens, emb, pos, h, hbf);

  for (int layer = 0; layer < NLAYERS; ++layer) {
    const __hip_bfloat16* W1l = W1bf + (size_t)layer * D_IN_PROJ_PAD * D_MODEL;
    const __hip_bfloat16* W2l = W2bf + (size_t)layer * D_MODEL * D_INNER;
    const float* cwl  = conv_w  + (size_t)layer * CONV_DIM * D_CONV;
    const float* cbl  = conv_b  + (size_t)layer * CONV_DIM;
    const float* dtbl = dt_bias + (size_t)layer * NHEADS;
    const float* all  = A_log   + (size_t)layer * NHEADS;
    const float* dskl = D_skip  + (size_t)layer * NHEADS;
    const float* nwl  = nrm_w   + (size_t)layer * D_INNER;
    const float* lnwl = ln_w    + (size_t)layer * D_MODEL;
    const float* lnbl = ln_b    + (size_t)layer * D_MODEL;

    // in_proj: zx[16384, 2368] = hbf @ W1^T (padded cols compute to zero)
    {
      int tilesM = ROWS / 32, tilesN = D_IN_PROJ_PAD / 64;
      int waves  = tilesM * tilesN;
      gemm_bf16_wmma<<<(waves + 7) / 8, 256, 0, stream>>>(
          reinterpret_cast<const __bf16*>(hbf),
          reinterpret_cast<const __bf16*>(W1l),
          zx, ROWS, D_IN_PROJ_PAD, D_MODEL);
    }

    conv_dt_kernel<<<ROWS, 256, 0, stream>>>(zx, cwl, cbl, dtbl, all, xbcs, dtb_, dAb);

    scan_kernel<<<BATCH * NHEADS, 256, 0, stream>>>(xbcs, dtb_, dAb, dskl, ybuf);

    gate_rms_kernel<<<ROWS, 256, 0, stream>>>(ybuf, zx, nwl, ybf);

    // out_proj: yprj[16384, 512] = ybf @ W2^T
    {
      int tilesM = ROWS / 32, tilesN = D_MODEL / 64;
      int waves  = tilesM * tilesN;
      gemm_bf16_wmma<<<(waves + 7) / 8, 256, 0, stream>>>(
          reinterpret_cast<const __bf16*>(ybf),
          reinterpret_cast<const __bf16*>(W2l),
          yprj, ROWS, D_MODEL, D_INNER);
    }

    resid_ln_kernel<<<ROWS, 256, 0, stream>>>(yprj, lnwl, lnbl, h, hbf);
  }

  pool_kernel<<<BATCH, 256, 0, stream>>>(h, pooled);
  head_kernel<<<BATCH, 128, 0, stream>>>(pooled, fc1_w, fc1_b, fc2_w, fc2_b, out);
}
